// TemporalGCN_71571335021033
// MI455X (gfx1250) — compile-verified, matrix operand order checked by reading, not run on hardware
//
#include <hip/hip_runtime.h>

// Problem constants (match reference)
#define B_   16
#define N_   512
#define T_   96
#define FIN  32
#define H_   128
#define FOUT 32
#define BN_  (B_*N_)                 // 8192
#define MROWS ((size_t)B_*N_*T_)     // 786432

typedef __attribute__((ext_vector_type(16))) __bf16 v16bf;
typedef __attribute__((ext_vector_type(8)))  float  v8f;

union ABFrag { v16bf v; unsigned int u[8]; };

__device__ __forceinline__ unsigned short f2bf(float f) {
  union { float f; unsigned int u; } x; x.f = f;
  unsigned int r = x.u + 0x7fffu + ((x.u >> 16) & 1u);   // RNE
  return (unsigned short)(r >> 16);
}
__device__ __forceinline__ float bf2f(unsigned short b) {
  union { unsigned int u; float f; } x; x.u = ((unsigned int)b) << 16; return x.f;
}
// 16-bit A/B fragment K index for element-pair v (0..7), lane half (0/1).
// Per ISA: lanes 0-15 hold K = [0..7 | 16..23], lanes 16-31 hold K = [8..15 | 24..31]
__device__ __forceinline__ int kmap(int v, int half) {
  return ((v < 4) ? 0 : 16) + half * 8 + ((v & 3) << 1);
}
__device__ __forceinline__ v8f vzero8() {
  v8f z; z[0]=0.f; z[1]=0.f; z[2]=0.f; z[3]=0.f; z[4]=0.f; z[5]=0.f; z[6]=0.f; z[7]=0.f;
  return z;
}

// ---------------------------------------------------------------------------
// CDNA5 async global->LDS copy (ASYNCcnt path, bypasses VGPRs).
// Per-lane: LDS[ldst] = MEM[gsrc] for 16 bytes.
// ---------------------------------------------------------------------------
__device__ __forceinline__ void async_copy_b128(const unsigned short* gsrc,
                                                unsigned short* ldst) {
  unsigned int lo = (unsigned int)(unsigned long long)(const void*)ldst; // LDS byte offset
  asm volatile("global_load_async_to_lds_b128 %0, %1, off"
               :: "v"(lo), "v"(gsrc) : "memory");
}
__device__ __forceinline__ void wait_async0() {
#if __has_builtin(__builtin_amdgcn_s_wait_asynccnt)
  __builtin_amdgcn_s_wait_asynccnt(0);
#else
  asm volatile("s_wait_asynccnt 0x0" ::: "memory");
#endif
}

// ---------------------------------------------------------------------------
// Small prep kernels
// ---------------------------------------------------------------------------
__global__ void k_rowsum_rsqrt(const float* __restrict__ adj, float* __restrict__ dinv) {
  int m = blockIdx.x * blockDim.x + threadIdx.x;
  if (m < N_) {
    float s = 1.0f;                       // identity contribution
    for (int n = 0; n < N_; ++n) s += adj[m * N_ + n];
    dinv[m] = (s > 0.0f) ? rsqrtf(s) : 0.0f;
  }
}

__global__ void k_normadj(const float* __restrict__ adj, const float* __restrict__ dinv,
                          unsigned short* __restrict__ adjb) {
  int i = blockIdx.x * 256 + threadIdx.x;       // N_*N_ = 262144
  int m = i >> 9, n = i & (N_ - 1);
  float v = adj[i] + ((m == n) ? 1.0f : 0.0f);
  adjb[i] = f2bf(dinv[m] * v * dinv[n]);
}

__global__ void k_mask_cvt(const float* __restrict__ x, unsigned short* __restrict__ xb,
                           size_t n) {
  size_t i = (size_t)blockIdx.x * 256 + threadIdx.x;
  if (i < n) {
    float v = x[i];
    if (v == -1.0f) v = 0.0f;                   // missing-value mask
    xb[i] = f2bf(v);
  }
}

__global__ void k_cvt(const float* __restrict__ a, unsigned short* __restrict__ b, int n) {
  int i = blockIdx.x * 256 + threadIdx.x;
  if (i < n) b[i] = f2bf(a[i]);
}

// in[R,C] row-major -> out[C,R] row-major, bf16
__global__ void k_cvtT(const float* __restrict__ a, unsigned short* __restrict__ b,
                       int R, int C) {
  int i = blockIdx.x * 256 + threadIdx.x;
  if (i < R * C) {
    int r = i / C, c = i % C;
    b[c * R + r] = f2bf(a[i]);
  }
}

__global__ void k_zero_bf16(unsigned short* __restrict__ p, int n) {
  int i = blockIdx.x * 256 + threadIdx.x;
  if (i < n) p[i] = 0;
}

// ---------------------------------------------------------------------------
// Generic bf16 WMMA GEMM:  out = act( A[M,K] * B[K,N] + bias )
// B is supplied COLUMN-MAJOR as Bt[N][K] (row-major, leading dim ldbT), so all
// LDS staging is straight async b128 copies (no in-kernel transpose scatter).
// 256 threads, 8 waves as 4(m) x 2(n); block tile 128x64; wave tile 32x32.
// Requires: M%128==0, N%64==0, K%32==0, lda/ldbT multiples of 8.
// OUTMODE 0: outB[row*ldc+col]
// OUTMODE 1: transposed "support" layout S_T[b][t][h=col][node], rows = [b][node][t]
// Bias index = col & bias_mask (compile-time specialized variants, no division).
// ---------------------------------------------------------------------------
#define GBM 128
#define GBN 64
#define GBK 32
#define GLA 40   // padded LDS strides (bf16 elements)
#define GLB 40

template <bool HAS_BIAS, bool RELU, int OUTMODE>
__global__ __launch_bounds__(256)
void k_gemm_bf16(const unsigned short* __restrict__ A,
                 const unsigned short* __restrict__ Bt,
                 const float* __restrict__ bias, int bias_mask,
                 unsigned short* __restrict__ outB,
                 int K, int lda, int ldbT, int ldc,
                 size_t sA, size_t sB, size_t sC) {
  __shared__ unsigned short a_s[GBM * GLA];
  __shared__ unsigned short b_s[GBN * GLB];

  const int tid  = threadIdx.x;
  const int lane = tid & 31;
  const int wave = tid >> 5;
  const int half = lane >> 4;
  const int l16  = lane & 15;
  const int wm   = wave >> 1;   // 0..3
  const int wn   = wave & 1;    // 0..1

  const size_t batch = blockIdx.z;
  A    += batch * sA;
  Bt   += batch * sB;
  outB += batch * sC;

  const int mBlock = blockIdx.y * GBM;
  const int nBlock = blockIdx.x * GBN;

  v8f c[2][2];
  c[0][0] = vzero8(); c[0][1] = vzero8(); c[1][0] = vzero8(); c[1][1] = vzero8();

  const int arow = tid >> 1, ako = (tid & 1) * 16;          // A: 16 bf16/thread
  const int bcol = tid >> 2, bko = (tid & 3) * 8;           // Bt: 8 bf16/thread

  for (int k0 = 0; k0 < K; k0 += GBK) {
    __syncthreads();
    {   // stage A: 128x32 via async copies
      const unsigned short* ga = A + (size_t)(mBlock + arow) * lda + k0 + ako;
      async_copy_b128(ga,     &a_s[arow * GLA + ako]);
      async_copy_b128(ga + 8, &a_s[arow * GLA + ako + 8]);
    }
    {   // stage Bt: 64 cols x 32 k via async copies (straight copy, already [n][k])
      async_copy_b128(Bt + (size_t)(nBlock + bcol) * ldbT + k0 + bko,
                      &b_s[bcol * GLB + bko]);
    }
    wait_async0();
    __syncthreads();

    ABFrag af[2], bf[2];
#pragma unroll
    for (int mt = 0; mt < 2; ++mt) {
      int row = wm * 32 + mt * 16 + l16;
#pragma unroll
      for (int v = 0; v < 8; ++v)
        af[mt].u[v] = *(const unsigned int*)&a_s[row * GLA + kmap(v, half)];
    }
#pragma unroll
    for (int nt = 0; nt < 2; ++nt) {
      int col = wn * 32 + nt * 16 + l16;
#pragma unroll
      for (int v = 0; v < 8; ++v)
        bf[nt].u[v] = *(const unsigned int*)&b_s[col * GLB + kmap(v, half)];
    }
#pragma unroll
    for (int mt = 0; mt < 2; ++mt)
#pragma unroll
      for (int nt = 0; nt < 2; ++nt)
        c[mt][nt] = __builtin_amdgcn_wmma_f32_16x16x32_bf16(
            false, af[mt].v, false, bf[nt].v, (short)0, c[mt][nt], false, false);
  }

  // epilogue: bias loaded once per n-tile, no per-element branches
  float bcv[2] = {0.0f, 0.0f};
  if (HAS_BIAS) {
#pragma unroll
    for (int nt = 0; nt < 2; ++nt)
      bcv[nt] = bias[(nBlock + wn * 32 + nt * 16 + l16) & bias_mask];
  }

#pragma unroll
  for (int mt = 0; mt < 2; ++mt)
#pragma unroll
    for (int nt = 0; nt < 2; ++nt)
#pragma unroll
      for (int i = 0; i < 8; ++i) {
        int row = mBlock + wm * 32 + mt * 16 + half * 8 + i;
        int col = nBlock + wn * 32 + nt * 16 + l16;
        float v = c[mt][nt][i];
        if (HAS_BIAS) v += bcv[nt];
        if (RELU) v = fmaxf(v, 0.0f);
        unsigned short bv = f2bf(v);
        if (OUTMODE == 0) {
          outB[(size_t)row * ldc + col] = bv;
        } else {
          int t    = row % T_;              // constants: strength-reduced
          int node = (row / T_) & (N_ - 1);
          int b    = row / (T_ * N_);
          outB[(((size_t)b * T_ + t) * H_ + col) * N_ + node] = bv;
        }
      }
}

// ---------------------------------------------------------------------------
// Fused GRU step + FC:
//   gh = h_prev[16,128] @ w_hh^T            (WMMA; whhB is bf16(w_hh) = [384n][128k])
//   gates (sigmoid/tanh) with precomputed gi slice, b_hh -> h_new
//   y = h_new[16,128] @ w_fc + b_fc         (wfcT = bf16(w_fc^T) = [32n][128k])
// Block: 256 threads, 16 rows; grid = BN/16 = 512 blocks.
// ---------------------------------------------------------------------------
union GruShm {
  unsigned short b[384 * 40];   // B-chunk stage [n][k], 30720 B
  float gh[16 * 392];           // gh result after GEMM, 25088 B
};

__global__ __launch_bounds__(256)
void k_gru_step(const unsigned short* __restrict__ gi,     // [BN*T, 384] bf16
                const unsigned short* __restrict__ hprev,  // [BN, 128]   bf16
                unsigned short* __restrict__ hnext,        // [BN, 128]   bf16
                const unsigned short* __restrict__ whhB,   // [384, 128]  bf16
                const unsigned short* __restrict__ wfcT,   // [32, 128]   bf16
                const float* __restrict__ b_hh,            // [384]
                const float* __restrict__ b_fc,            // [32]
                float* __restrict__ out,                   // [BN, T, 32]
                int t) {
  __shared__ unsigned short a_s[16 * 136];     // h_prev tile
  __shared__ GruShm sm;
  __shared__ unsigned short hnew_s[16 * 136];
  __shared__ unsigned short wfc_s[32 * 136];   // wfc^T [n][k]

  const int tid  = threadIdx.x;
  const int lane = tid & 31;
  const int wave = tid >> 5;
  const int half = lane >> 4;
  const int l16  = lane & 15;
  const int rowBlock = blockIdx.x * 16;

  {   // stage A (h_prev 16x128): async 8 bf16 per thread
    int row = tid >> 4, ko = (tid & 15) * 8;
    async_copy_b128(hprev + (size_t)(rowBlock + row) * H_ + ko,
                    &a_s[row * 136 + ko]);
  }

  v8f c[3];
  c[0] = vzero8(); c[1] = vzero8(); c[2] = vzero8();

  for (int kc = 0; kc < 128; kc += 32) {
    __syncthreads();
    // stage B chunk [384 n][32 k] via async straight copies
#pragma unroll
    for (int j = 0; j < 6; ++j) {
      int g = tid + j * 256;                  // 0..1535
      int col = g >> 2, k8 = (g & 3) * 8;
      async_copy_b128(whhB + (size_t)col * H_ + kc + k8, &sm.b[col * 40 + k8]);
    }
    wait_async0();   // first iteration also covers the h_prev stage
    __syncthreads();

    ABFrag af;
#pragma unroll
    for (int v = 0; v < 8; ++v)
      af.u[v] = *(const unsigned int*)&a_s[l16 * 136 + kc + kmap(v, half)];
#pragma unroll
    for (int nt = 0; nt < 3; ++nt) {
      int col = wave * 48 + nt * 16 + l16;
      ABFrag bf;
#pragma unroll
      for (int v = 0; v < 8; ++v)
        bf.u[v] = *(const unsigned int*)&sm.b[col * 40 + kmap(v, half)];
      c[nt] = __builtin_amdgcn_wmma_f32_16x16x32_bf16(
          false, af.v, false, bf.v, (short)0, c[nt], false, false);
    }
  }
  __syncthreads();   // all waves done reading sm.b -> safe to overwrite with sm.gh

#pragma unroll
  for (int nt = 0; nt < 3; ++nt)
#pragma unroll
    for (int i = 0; i < 8; ++i) {
      int row = half * 8 + i;
      int col = wave * 48 + nt * 16 + l16;
      sm.gh[row * 392 + col] = c[nt][i];
    }
  __syncthreads();

  // Gate phase: 16 x 128 = 2048 elements, 8 per thread
#pragma unroll
  for (int j = 0; j < 8; ++j) {
    int idx = tid + j * 256;
    int r = idx >> 7, hc = idx & 127;
    const unsigned short* gp = gi + ((size_t)(rowBlock + r) * T_ + t) * 384;
    float gir = bf2f(gp[hc]);
    float giz = bf2f(gp[hc + 128]);
    float gin = bf2f(gp[hc + 256]);
    float ghr = sm.gh[r * 392 + hc]       + b_hh[hc];
    float ghz = sm.gh[r * 392 + hc + 128] + b_hh[hc + 128];
    float ghn = sm.gh[r * 392 + hc + 256] + b_hh[hc + 256];
    float hp  = bf2f(a_s[r * 136 + hc]);
    float rg = 1.0f / (1.0f + __expf(-(gir + ghr)));
    float zg = 1.0f / (1.0f + __expf(-(giz + ghz)));
    float ng = tanhf(gin + rg * ghn);
    float hn = (1.0f - zg) * ng + zg * hp;
    unsigned short hb = f2bf(hn);
    hnew_s[r * 136 + hc] = hb;
    hnext[(size_t)(rowBlock + r) * H_ + hc] = hb;
  }

  {   // stage wfc^T [32 n][128 k] via async straight copies
#pragma unroll
    for (int j = 0; j < 2; ++j) {
      int g = tid + j * 256;                  // 0..511
      int col = g >> 4, k8 = (g & 15) * 8;
      async_copy_b128(wfcT + (size_t)col * H_ + k8, &wfc_s[col * 136 + k8]);
    }
  }
  wait_async0();
  __syncthreads();

  // FC: y[16,32] = h_new[16,128] @ wfc[128,32]; waves 0,1 each take 16 cols
  if (wave < 2) {
    v8f y = vzero8();
    for (int kc = 0; kc < 128; kc += 32) {
      ABFrag af, bf;
#pragma unroll
      for (int v = 0; v < 8; ++v) {
        int kv = kc + kmap(v, half);
        af.u[v] = *(const unsigned int*)&hnew_s[l16 * 136 + kv];
        bf.u[v] = *(const unsigned int*)&wfc_s[(wave * 16 + l16) * 136 + kv];
      }
      y = __builtin_amdgcn_wmma_f32_16x16x32_bf16(
          false, af.v, false, bf.v, (short)0, y, false, false);
    }
#pragma unroll
    for (int i = 0; i < 8; ++i) {
      int row = rowBlock + half * 8 + i;
      int col = wave * 16 + l16;
      out[(size_t)row * (T_ * FOUT) + (size_t)t * FOUT + col] = y[i] + b_fc[col];
    }
  }
}

// ---------------------------------------------------------------------------
// Host launcher
// ---------------------------------------------------------------------------
static inline unsigned char* ws_take(unsigned char*& p, size_t bytes) {
  unsigned char* r = p;
  p += (bytes + 255) & ~(size_t)255;
  return r;
}

extern "C" void kernel_launch(void* const* d_in, const int* in_sizes, int n_in,
                              void* d_out, int out_size, void* d_ws, size_t ws_size,
                              hipStream_t stream) {
  (void)in_sizes; (void)n_in; (void)out_size; (void)ws_size;

  const float* x     = (const float*)d_in[0];   // [B,N,T,FIN]
  const float* adj   = (const float*)d_in[1];   // [N,N]
  const float* gc_w0 = (const float*)d_in[2];   // [FIN,H]
  const float* gc_b0 = (const float*)d_in[3];   // [H]
  const float* gc_w1 = (const float*)d_in[4];   // [H,H]
  const float* gc_b1 = (const float*)d_in[5];   // [H]
  const float* w_ih  = (const float*)d_in[6];   // [3H,H]
  const float* w_hh  = (const float*)d_in[7];   // [3H,H]
  const float* b_ih  = (const float*)d_in[8];   // [3H]
  const float* b_hh  = (const float*)d_in[9];   // [3H]
  const float* w_fc  = (const float*)d_in[10];  // [H,FOUT]
  const float* b_fc  = (const float*)d_in[11];  // [FOUT]
  float* out = (float*)d_out;

  unsigned char* wp = (unsigned char*)d_ws;
  float*          dinv = (float*)         ws_take(wp, N_ * sizeof(float));
  unsigned short* adjb = (unsigned short*)ws_take(wp, (size_t)N_ * N_ * 2);
  unsigned short* w0T  = (unsigned short*)ws_take(wp, (size_t)FIN * H_ * 2);  // [H][FIN]
  unsigned short* w1T  = (unsigned short*)ws_take(wp, (size_t)H_ * H_ * 2);   // [H][H]
  unsigned short* wihB = (unsigned short*)ws_take(wp, (size_t)3 * H_ * H_ * 2); // [3H][H]
  unsigned short* whhB = (unsigned short*)ws_take(wp, (size_t)3 * H_ * H_ * 2); // [3H][H]
  unsigned short* wfcT = (unsigned short*)ws_take(wp, (size_t)H_ * FOUT * 2); // [FOUT][H]
  unsigned short* xb   = (unsigned short*)ws_take(wp, MROWS * FIN * 2);
  unsigned short* Sbuf = (unsigned short*)ws_take(wp, MROWS * H_ * 2);  // S_T [b][t][h][node]
  unsigned short* Hbuf = (unsigned short*)ws_take(wp, MROWS * H_ * 2);  // [b][node][t][h]
  unsigned short* GI   = (unsigned short*)ws_take(wp, MROWS * 3 * H_ * 2);
  unsigned short* hA   = (unsigned short*)ws_take(wp, (size_t)BN_ * H_ * 2);
  unsigned short* hB   = (unsigned short*)ws_take(wp, (size_t)BN_ * H_ * 2);

  // 1) adjacency normalization
  k_rowsum_rsqrt<<<2, 256, 0, stream>>>(adj, dinv);
  k_normadj<<<(N_ * N_) / 256, 256, 0, stream>>>(adj, dinv, adjb);

  // 2) weight conversions (all B operands stored column-major [n][k])
  k_cvtT<<<(FIN * H_ + 255) / 256, 256, 0, stream>>>(gc_w0, w0T, FIN, H_);
  k_cvtT<<<(H_ * H_ + 255) / 256, 256, 0, stream>>>(gc_w1, w1T, H_, H_);
  k_cvt <<<(3 * H_ * H_ + 255) / 256, 256, 0, stream>>>(w_ih, wihB, 3 * H_ * H_);
  k_cvt <<<(3 * H_ * H_ + 255) / 256, 256, 0, stream>>>(w_hh, whhB, 3 * H_ * H_);
  k_cvtT<<<(H_ * FOUT + 255) / 256, 256, 0, stream>>>(w_fc, wfcT, H_, FOUT);

  // 3) mask + convert x
  {
    size_t n = MROWS * FIN;
    k_mask_cvt<<<(unsigned)((n + 255) / 256), 256, 0, stream>>>(x, xb, n);
  }

  const size_t aggStride = (size_t)N_ * T_ * H_;   // 512*12288

  // 4) GCN layer 0: S_T = (xb @ W0) in transposed layout ; H = relu(adjb @ S + b0)
  k_gemm_bf16<false, false, 1>
      <<<dim3(H_ / GBN, (unsigned)(MROWS / GBM), 1), 256, 0, stream>>>(
      xb, w0T, nullptr, 0, Sbuf, FIN, FIN, FIN, 0, 0, 0, 0);
  k_gemm_bf16<true, true, 0>
      <<<dim3((T_ * H_) / GBN, N_ / GBM, B_), 256, 0, stream>>>(
      adjb, Sbuf, gc_b0, H_ - 1, Hbuf, N_, N_, N_, T_ * H_,
      0, aggStride, aggStride);

  // 5) GCN layer 1
  k_gemm_bf16<false, false, 1>
      <<<dim3(H_ / GBN, (unsigned)(MROWS / GBM), 1), 256, 0, stream>>>(
      Hbuf, w1T, nullptr, 0, Sbuf, H_, H_, H_, 0, 0, 0, 0);
  k_gemm_bf16<true, true, 0>
      <<<dim3((T_ * H_) / GBN, N_ / GBM, B_), 256, 0, stream>>>(
      adjb, Sbuf, gc_b1, H_ - 1, Hbuf, N_, N_, N_, T_ * H_,
      0, aggStride, aggStride);

  // 6) time-independent GRU input gates: GI = H @ w_ih^T + b_ih
  //    cols < 384, so mask 511 is the identity (no modulo needed)
  k_gemm_bf16<true, false, 0>
      <<<dim3((3 * H_) / GBN, (unsigned)(MROWS / GBM), 1), 256, 0, stream>>>(
      Hbuf, wihB, b_ih, 511, GI, H_, H_, H_, 3 * H_, 0, 0, 0);

  // 7) GRU recurrence (h0 = 0), fused gates + FC per step
  k_zero_bf16<<<(BN_ * H_) / 256, 256, 0, stream>>>(hA, BN_ * H_);
  for (int t = 0; t < T_; ++t) {
    unsigned short* hin  = (t & 1) ? hB : hA;
    unsigned short* hout = (t & 1) ? hA : hB;
    k_gru_step<<<BN_ / 16, 256, 0, stream>>>(GI, hin, hout, whhB, wfcT,
                                             b_hh, b_fc, out, t);
  }
}
